// GCRNN_19499151524295
// MI455X (gfx1250) — matrix operand order repaired and checked by reading, not compiled, not run on hardware
//
#include <hip/hip_runtime.h>
#include <hip/hip_bf16.h>

// ---------------------------------------------------------------------------
// Types for CDNA5 WMMA (wave32, 16x16x32 bf16 -> f32)
// ---------------------------------------------------------------------------
typedef __attribute__((ext_vector_type(16))) __bf16 v16bf;
typedef __attribute__((ext_vector_type(8)))  __bf16 v8bf;
typedef __attribute__((ext_vector_type(8)))  float  v8f;

__device__ __forceinline__ __bf16 f32_to_bf16(float f) {
  union { float f; unsigned u; } c; c.f = f;
  unsigned r = (c.u + 0x7FFFu + ((c.u >> 16) & 1u)) >> 16;
  unsigned short s = (unsigned short)r;
  __bf16 out;
  __builtin_memcpy(&out, &s, 2);
  return out;
}

// A fragment, contiguous k order used here (not the ISA table order), matching B:
// lanes 0-15: K = 0..15; lanes 16-31: K = 16..31
__device__ __forceinline__ v16bf load_frag_a(const __bf16* base) {
  v8bf lo = *(const v8bf*)(base);
  v8bf hi = *(const v8bf*)(base + 8);
  v16bf r;
#pragma unroll
  for (int i = 0; i < 8; ++i) { r[i] = lo[i]; r[i + 8] = hi[i]; }
  return r;
}

// B fragment: lane n = lane&15, lanes 0-15 hold K=0..15, lanes 16-31 K=16..31
__device__ __forceinline__ v16bf load_frag_b(const __bf16* base) {
  v8bf lo = *(const v8bf*)(base);
  v8bf hi = *(const v8bf*)(base + 8);
  v16bf r;
#pragma unroll
  for (int i = 0; i < 8; ++i) { r[i] = lo[i]; r[i + 8] = hi[i]; }
  return r;
}

// ---------------------------------------------------------------------------
// Generic WMMA GEMM:  C[M,Nout] = act( A[M,K]@W + (A2[M,K]@W2) + bias1 + bias2 )
// Wt is the transposed weight, [Nout, K] bf16, so B fragments load contiguously.
// Block = 256 threads = 8 waves; block tile 128x64; wave tile 16x64.
// ---------------------------------------------------------------------------
__global__ __launch_bounds__(256)
void gemm_wmma_bf16(const __bf16* __restrict__ A,  const __bf16* __restrict__ Wt,
                    const __bf16* __restrict__ A2, const __bf16* __restrict__ Wt2,
                    const float* __restrict__ bias1, const float* __restrict__ bias2,
                    float* __restrict__ C, int M, int K, int Nout, int relu_flag) {
  const int lane = threadIdx.x & 31;
  const int wave = threadIdx.x >> 5;
  const int mbase = blockIdx.x * 128 + wave * 16;
  const int nbase = blockIdx.y * 64;
  const int half = (lane >> 4) & 1;
  const int lrow = lane & 15;

  int arow = mbase + lrow;
  if (arow >= M) arow = M - 1;                    // clamp; stores are predicated

  v8f acc[4] = {};

  const __bf16* aptr = A + (size_t)arow * K + half * 16;
  const __bf16* bptr0 = Wt + (size_t)(nbase + 0 * 16 + lrow) * K + half * 16;
  const __bf16* bptr1 = Wt + (size_t)(nbase + 1 * 16 + lrow) * K + half * 16;
  const __bf16* bptr2 = Wt + (size_t)(nbase + 2 * 16 + lrow) * K + half * 16;
  const __bf16* bptr3 = Wt + (size_t)(nbase + 3 * 16 + lrow) * K + half * 16;

  for (int kt = 0; kt < K; kt += 32) {
    v16bf a = load_frag_a(aptr + kt);
    acc[0] = __builtin_amdgcn_wmma_f32_16x16x32_bf16(false, a, false, load_frag_b(bptr0 + kt), (short)0, acc[0], false, false);
    acc[1] = __builtin_amdgcn_wmma_f32_16x16x32_bf16(false, a, false, load_frag_b(bptr1 + kt), (short)0, acc[1], false, false);
    acc[2] = __builtin_amdgcn_wmma_f32_16x16x32_bf16(false, a, false, load_frag_b(bptr2 + kt), (short)0, acc[2], false, false);
    acc[3] = __builtin_amdgcn_wmma_f32_16x16x32_bf16(false, a, false, load_frag_b(bptr3 + kt), (short)0, acc[3], false, false);
  }

  if (A2 != nullptr) {                            // fused second GEMM, same K
    const __bf16* a2ptr = A2 + (size_t)arow * K + half * 16;
    const __bf16* cptr0 = Wt2 + (size_t)(nbase + 0 * 16 + lrow) * K + half * 16;
    const __bf16* cptr1 = Wt2 + (size_t)(nbase + 1 * 16 + lrow) * K + half * 16;
    const __bf16* cptr2 = Wt2 + (size_t)(nbase + 2 * 16 + lrow) * K + half * 16;
    const __bf16* cptr3 = Wt2 + (size_t)(nbase + 3 * 16 + lrow) * K + half * 16;
    for (int kt = 0; kt < K; kt += 32) {
      v16bf a = load_frag_a(a2ptr + kt);
      acc[0] = __builtin_amdgcn_wmma_f32_16x16x32_bf16(false, a, false, load_frag_b(cptr0 + kt), (short)0, acc[0], false, false);
      acc[1] = __builtin_amdgcn_wmma_f32_16x16x32_bf16(false, a, false, load_frag_b(cptr1 + kt), (short)0, acc[1], false, false);
      acc[2] = __builtin_amdgcn_wmma_f32_16x16x32_bf16(false, a, false, load_frag_b(cptr2 + kt), (short)0, acc[2], false, false);
      acc[3] = __builtin_amdgcn_wmma_f32_16x16x32_bf16(false, a, false, load_frag_b(cptr3 + kt), (short)0, acc[3], false, false);
    }
  }

  // Epilogue: C/D layout — lane n=lane&15, VGPR r -> M = r + 8*half
  const int mrow0 = mbase + half * 8;
#pragma unroll
  for (int j = 0; j < 4; ++j) {
    const int col = nbase + j * 16 + lrow;
    float bsum = 0.f;
    if (bias1) bsum += bias1[col];
    if (bias2) bsum += bias2[col];
#pragma unroll
    for (int r = 0; r < 8; ++r) {
      const int row = mrow0 + r;
      if (row < M) {
        float v = acc[j][r] + bsum;
        if (relu_flag) v = fmaxf(v, 0.f);
        C[(size_t)row * Nout + col] = v;
      }
    }
  }
}

// ---------------------------------------------------------------------------
// Small utility kernels
// ---------------------------------------------------------------------------
__global__ void k_zero_f32(float* p, long n) {
  long i = (long)blockIdx.x * blockDim.x + threadIdx.x;
  if (i < n) p[i] = 0.f;
}

__global__ void k_f32_to_bf16(const float* __restrict__ s, __bf16* __restrict__ d, long n) {
  long i = (long)blockIdx.x * blockDim.x + threadIdx.x;
  if (i < n) d[i] = f32_to_bf16(s[i]);
}

// W[K,Nout] f32 -> Wt[Nout,K] bf16
__global__ void k_transpose_bf16(const float* __restrict__ W, __bf16* __restrict__ Wt,
                                 int K, int Nout) {
  long i = (long)blockIdx.x * blockDim.x + threadIdx.x;
  if (i < (long)K * Nout) {
    int k = (int)(i / Nout), n = (int)(i % Nout);
    Wt[(size_t)n * K + k] = f32_to_bf16(W[i]);
  }
}

__global__ void k_deg_scatter(const int* __restrict__ dst, const float* __restrict__ w,
                              float* __restrict__ deg, int E) {
  int e = blockIdx.x * blockDim.x + threadIdx.x;
  if (e < E) atomicAdd(&deg[dst[e]], w[e]);
}

__global__ void k_dinv(float* deg, int N) {            // self-loop weight 1.0 folded in
  int i = blockIdx.x * blockDim.x + threadIdx.x;
  if (i < N) deg[i] = rsqrtf(deg[i] + 1.0f);
}

__global__ void k_norm(const int* __restrict__ src, const int* __restrict__ dst,
                       const float* __restrict__ w, const float* __restrict__ dinv,
                       float* __restrict__ norm, int E) {
  int e = blockIdx.x * blockDim.x + threadIdx.x;
  if (e < E) norm[e] = dinv[src[e]] * w[e] * dinv[dst[e]];
}

// One wave per edge; lanes sweep the feature dimension (coalesced 128B chunks).
__global__ __launch_bounds__(256)
void k_scatter_edges(const float* __restrict__ T, const int* __restrict__ src,
                     const int* __restrict__ dst, const float* __restrict__ norm,
                     float* __restrict__ out, int E, int F) {
  int wv = (blockIdx.x * blockDim.x + threadIdx.x) >> 5;
  int lane = threadIdx.x & 31;
  if (wv >= E) return;
  int s = src[wv], d = dst[wv];
  float nr = norm[wv];
  const float* trow = T + (size_t)s * F;
  float* orow = out + (size_t)d * F;
  for (int f = lane; f < F; f += 32) atomicAdd(&orow[f], trow[f] * nr);
}

// f = relu(agg + selfloop + b1); G = f + prev_h  (bf16 for next GEMM)
__global__ void k_finalize1(const float* __restrict__ F, const float* __restrict__ T1,
                            const float* __restrict__ dinv, const float* __restrict__ b1,
                            const float* __restrict__ prevh, __bf16* __restrict__ Gbf, int N) {
  long idx = (long)blockIdx.x * blockDim.x + threadIdx.x;
  if (idx < (long)N * 256) {
    int n = (int)(idx >> 8), d = (int)(idx & 255);
    float di = dinv[n];
    float g = fmaxf(F[idx] + T1[idx] * di * di + b1[d], 0.f) + prevh[idx];
    Gbf[idx] = f32_to_bf16(g);
  }
}

// z = relu(agg2 + selfloop + b2); write z into XZ[:, :128] (bf16) and pooled sum
__global__ __launch_bounds__(256)
void k_finalize2(const float* __restrict__ F2, const float* __restrict__ T2,
                 const float* __restrict__ dinv, const float* __restrict__ b2,
                 __bf16* __restrict__ XZ, float* __restrict__ zsum, int N) {
  __shared__ float part[128];
  const int d = threadIdx.x & 127;
  const int sub = threadIdx.x >> 7;                // 0/1
  if (threadIdx.x < 128) part[threadIdx.x] = 0.f;
  __syncthreads();
  const int NPB = 32;
  const int n0 = blockIdx.x * NPB;
  float local = 0.f;
  for (int i = sub; i < NPB; i += 2) {
    int n = n0 + i;
    if (n < N) {
      float di = dinv[n];
      size_t off = (size_t)n * 128 + d;
      float zv = fmaxf(F2[off] + T2[off] * di * di + b2[d], 0.f);
      XZ[(size_t)n * 256 + d] = f32_to_bf16(zv);
      local += zv;
    }
  }
  atomicAdd(&part[d], local);
  __syncthreads();
  if (threadIdx.x < 128) atomicAdd(&zsum[threadIdx.x], part[threadIdx.x]);
}

// copy x (bf16) into XZ[:, 128:256]
__global__ void k_x_to_xz(const float* __restrict__ x, __bf16* __restrict__ XZ, int N) {
  long idx = (long)blockIdx.x * blockDim.x + threadIdx.x;
  if (idx < (long)N * 128) {
    int n = (int)(idx >> 7), d = (int)(idx & 127);
    XZ[(size_t)n * 256 + 128 + d] = f32_to_bf16(x[idx]);
  }
}

__global__ void k_head(const float* __restrict__ zsum, const float* __restrict__ Wh,
                       const float* __restrict__ bh, float* __restrict__ out, int N) {
  __shared__ float s[128];
  int d = threadIdx.x;
  s[d] = (zsum[d] / (float)N) * Wh[d];
  __syncthreads();
  for (int off = 64; off > 0; off >>= 1) {
    if (d < off) s[d] += s[d + off];
    __syncthreads();
  }
  if (d == 0) out[0] = s[0] + bh[0];
}

// R = sigmoid(Ur); S = bf16(prev_h * R)
__global__ void k_gru_rs(const float* __restrict__ Ur, const float* __restrict__ prevh,
                         __bf16* __restrict__ Sbf, long n) {
  long i = (long)blockIdx.x * blockDim.x + threadIdx.x;
  if (i < n) {
    float r = 1.f / (1.f + __expf(-Ur[i]));
    Sbf[i] = f32_to_bf16(prevh[i] * r);
  }
}

// h = Z*prev + (1-Z)*tanh(Uh), Z = sigmoid(Uz)
__global__ void k_gru_final(const float* __restrict__ Uz, const float* __restrict__ Uh,
                            const float* __restrict__ prevh, float* __restrict__ hout, long n) {
  long i = (long)blockIdx.x * blockDim.x + threadIdx.x;
  if (i < n) {
    float z = 1.f / (1.f + __expf(-Uz[i]));
    float ht = tanhf(Uh[i]);
    hout[i] = z * prevh[i] + (1.f - z) * ht;
  }
}

// ---------------------------------------------------------------------------
// Launch
// ---------------------------------------------------------------------------
extern "C" void kernel_launch(void* const* d_in, const int* in_sizes, int n_in,
                              void* d_out, int out_size, void* d_ws, size_t ws_size,
                              hipStream_t stream) {
  const int DIN = 128, H = 256;
  const int N = in_sizes[0] / DIN;
  const int E = in_sizes[2];

  const float* x      = (const float*)d_in[0];
  const int*   eidx   = (const int*)d_in[1];
  const int*   esrc   = eidx;
  const int*   edst   = eidx + E;
  const float* ew     = (const float*)d_in[2];
  const float* prevh  = (const float*)d_in[3];
  const float* W1 = (const float*)d_in[4],  *b1  = (const float*)d_in[5];
  const float* W2 = (const float*)d_in[6],  *b2  = (const float*)d_in[7];
  const float* Wxz = (const float*)d_in[8], *bxz = (const float*)d_in[9];
  const float* Whz = (const float*)d_in[10],*bhz = (const float*)d_in[11];
  const float* Wxr = (const float*)d_in[12],*bxr = (const float*)d_in[13];
  const float* Whr = (const float*)d_in[14],*bhr = (const float*)d_in[15];
  const float* Wxh = (const float*)d_in[16],*bxh = (const float*)d_in[17];
  const float* Whh = (const float*)d_in[18],*bhh = (const float*)d_in[19];
  const float* Whead = (const float*)d_in[20], *bhead = (const float*)d_in[21];
  float* out = (float*)d_out;

  // bump allocator over d_ws
  char* ws = (char*)d_ws;
  size_t off = 0;
  auto alloc = [&](size_t bytes) -> void* {
    off = (off + 255) & ~(size_t)255;
    void* p = ws + off;
    off += bytes;
    return p;
  };

  float* deg  = (float*)alloc((size_t)N * 4);          // becomes dinv in place
  float* norm = (float*)alloc((size_t)E * 4);
  float* T1   = (float*)alloc((size_t)N * H * 4);      // reused as Uz
  float* F    = (float*)alloc((size_t)N * H * 4);      // reused as Ur
  float* TF2  = (float*)alloc((size_t)N * H * 4);      // [T2 | F2] then reused as Uh
  float* T2   = TF2;
  float* F2   = TF2 + (size_t)N * 128;
  __bf16* Xbf  = (__bf16*)alloc((size_t)N * DIN * 2);
  __bf16* Gbf  = (__bf16*)alloc((size_t)N * H * 2);    // reused as Sbf
  __bf16* XZbf = (__bf16*)alloc((size_t)N * H * 2);
  __bf16* PHbf = (__bf16*)alloc((size_t)N * H * 2);
  __bf16* W1t  = (__bf16*)alloc((size_t)DIN * H * 2);  // [256 x 128]
  __bf16* W2t  = (__bf16*)alloc((size_t)H * 128 * 2);  // [128 x 256]
  __bf16* Wxzt = (__bf16*)alloc((size_t)H * H * 2);
  __bf16* Whzt = (__bf16*)alloc((size_t)H * H * 2);
  __bf16* Wxrt = (__bf16*)alloc((size_t)H * H * 2);
  __bf16* Whrt = (__bf16*)alloc((size_t)H * H * 2);
  __bf16* Wxht = (__bf16*)alloc((size_t)H * H * 2);
  __bf16* Whht = (__bf16*)alloc((size_t)H * H * 2);
  float* zsum = (float*)alloc(128 * 4);
  float* Uz = T1; float* Ur = F; float* Uh = TF2; __bf16* Sbf = Gbf;

  auto cdiv = [](long a, long b) { return (int)((a + b - 1) / b); };
  const long NH = (long)N * H;

  // ---- degree / norm --------------------------------------------------------
  k_zero_f32<<<cdiv(N, 256), 256, 0, stream>>>(deg, N);
  k_deg_scatter<<<cdiv(E, 256), 256, 0, stream>>>(edst, ew, deg, E);
  k_dinv<<<cdiv(N, 256), 256, 0, stream>>>(deg, N);
  k_norm<<<cdiv(E, 256), 256, 0, stream>>>(esrc, edst, ew, deg, norm, E);

  // ---- bf16 conversions / weight transposes ---------------------------------
  k_f32_to_bf16<<<cdiv((long)N * DIN, 256), 256, 0, stream>>>(x, Xbf, (long)N * DIN);
  k_f32_to_bf16<<<cdiv(NH, 256), 256, 0, stream>>>(prevh, PHbf, NH);
  k_x_to_xz<<<cdiv((long)N * DIN, 256), 256, 0, stream>>>(x, XZbf, N);
  k_transpose_bf16<<<cdiv((long)DIN * H, 256), 256, 0, stream>>>(W1, W1t, DIN, H);
  k_transpose_bf16<<<cdiv((long)H * 128, 256), 256, 0, stream>>>(W2, W2t, H, 128);
  k_transpose_bf16<<<cdiv((long)H * H, 256), 256, 0, stream>>>(Wxz, Wxzt, H, H);
  k_transpose_bf16<<<cdiv((long)H * H, 256), 256, 0, stream>>>(Whz, Whzt, H, H);
  k_transpose_bf16<<<cdiv((long)H * H, 256), 256, 0, stream>>>(Wxr, Wxrt, H, H);
  k_transpose_bf16<<<cdiv((long)H * H, 256), 256, 0, stream>>>(Whr, Whrt, H, H);
  k_transpose_bf16<<<cdiv((long)H * H, 256), 256, 0, stream>>>(Wxh, Wxht, H, H);
  k_transpose_bf16<<<cdiv((long)H * H, 256), 256, 0, stream>>>(Whh, Whht, H, H);

  dim3 blk(256);

  // ---- GCN layer 1: T1 = x @ W1 ; F = agg(T1) ; G = relu(F + b1) + prev_h ----
  {
    dim3 grid(cdiv(N, 128), H / 64);
    gemm_wmma_bf16<<<grid, blk, 0, stream>>>(Xbf, W1t, nullptr, nullptr,
                                             nullptr, nullptr, T1, N, DIN, H, 0);
  }
  k_zero_f32<<<cdiv(NH, 256), 256, 0, stream>>>(F, NH);
  k_scatter_edges<<<cdiv((long)E * 32, 256), 256, 0, stream>>>(T1, esrc, edst, norm, F, E, H);
  k_finalize1<<<cdiv(NH, 256), 256, 0, stream>>>(F, T1, deg, b1, prevh, Gbf, N);

  // ---- GCN layer 2: T2 = G @ W2 ; z = relu(agg(T2)+b2) ; mean-pool ----------
  {
    dim3 grid(cdiv(N, 128), 128 / 64);
    gemm_wmma_bf16<<<grid, blk, 0, stream>>>(Gbf, W2t, nullptr, nullptr,
                                             nullptr, nullptr, T2, N, H, 128, 0);
  }
  k_zero_f32<<<cdiv((long)N * 128, 256), 256, 0, stream>>>(F2, (long)N * 128);
  k_zero_f32<<<1, 128, 0, stream>>>(zsum, 128);
  k_scatter_edges<<<cdiv((long)E * 32, 256), 256, 0, stream>>>(T2, esrc, edst, norm, F2, E, 128);
  k_finalize2<<<cdiv(N, 32), 256, 0, stream>>>(F2, T2, deg, b2, XZbf, zsum, N);
  k_head<<<1, 128, 0, stream>>>(zsum, Whead, bhead, out, N);

  // ---- GRU: fused dual GEMMs (XZ@Wx* + H@Wh*) -------------------------------
  {
    dim3 grid(cdiv(N, 128), H / 64);
    gemm_wmma_bf16<<<grid, blk, 0, stream>>>(XZbf, Wxzt, PHbf, Whzt, bxz, bhz, Uz, N, H, H, 0);
    gemm_wmma_bf16<<<grid, blk, 0, stream>>>(XZbf, Wxrt, PHbf, Whrt, bxr, bhr, Ur, N, H, H, 0);
    k_gru_rs<<<cdiv(NH, 256), 256, 0, stream>>>(Ur, prevh, Sbf, NH);
    gemm_wmma_bf16<<<grid, blk, 0, stream>>>(XZbf, Wxht, Sbf, Whht, bxh, bhh, Uh, N, H, H, 0);
  }
  k_gru_final<<<cdiv(NH, 256), 256, 0, stream>>>(Uz, Uh, prevh, out + 1, NH);

  (void)n_in; (void)out_size; (void)ws_size;
}
